// MultimodalP2MFDS_Final_Diffusion_53987738910751
// MI455X (gfx1250) — compile-verified, matrix operand order checked by reading, not run on hardware
//
#include <hip/hip_runtime.h>
#include <hip/hip_bf16.h>
#include <math.h>

typedef __bf16 bf16_t;
typedef bf16_t v16bf __attribute__((ext_vector_type(16)));
typedef float  v8f   __attribute__((ext_vector_type(8)));

// pack two fp32 into packed bf16x2: round-to-nearest via +0x8000, then a single
// v_perm_b32 extracts both high halves.
static __device__ __forceinline__ unsigned pack2bf(float f0, float f1) {
    unsigned u0 = __builtin_bit_cast(unsigned, f0) + 0x8000u;
    unsigned u1 = __builtin_bit_cast(unsigned, f1) + 0x8000u;
    return __builtin_amdgcn_perm(u1, u0, 0x07060302u);  // {u1.b3,u1.b2,u0.b3,u0.b2}
}

static __device__ __forceinline__ unsigned long long pack4bf(float x, float y, float z, float w) {
    return (unsigned long long)pack2bf(x, y)
         | ((unsigned long long)pack2bf(z, w) << 32);
}

// k-position permutation: swap bits 3<->4 so each lane's 16 fragment elements
// (ISA 7.12.2: k = h*8+{0..7}, h*8+{16..23}) are contiguous in LDS.
static __device__ __forceinline__ int kpos(int k) {
    return (k & 7) | ((k & 8) << 1) | ((k & 16) >> 1);
}

// ---------------------------------------------------------------------------
// Generic WMMA GEMM:  C[M,N] = epilogue( A[M,K] @ W[K,N] + bias )
// Block = 256 threads (8 waves). Block tile 64(M) x 64(N), k-step 32.
// Register-pipelined staging (global loads for tile k+1 issued during step k),
// double-buffered LDS, ONE barrier per k-step. Each wave: 1 A fragment,
// 2 B fragments, 2 x v_wmma_f32_16x16x32_bf16 (A operand reused).
// EPI: 0 = store (+bias), 1 = exact GELU, 2 = top1-masked store (MoE e==0),
//      3 = top1-masked accumulate (MoE e>0)
// ---------------------------------------------------------------------------
template <int EPI>
__global__ __launch_bounds__(256) void gemm_wmma(
    const float* __restrict__ A, const float* __restrict__ W,
    const float* __restrict__ bias, float* __restrict__ C,
    int M, int N, int K, const int* __restrict__ top1, int expert) {
    __shared__ alignas(16) unsigned short As[2][64][40];
    __shared__ alignas(16) unsigned short Bs[2][64][40];

    const int tid  = threadIdx.x;
    const int m0   = blockIdx.y * 64;
    const int n0   = blockIdx.x * 64;
    const int w    = tid >> 5;
    const int lane = tid & 31;
    const int wm   = w & 3;          // 4 M-tiles
    const int wn   = w >> 2;         // 2 N-pairs (32 cols each)
    const int half = lane >> 4;
    const int l16  = lane & 15;

    // staging assignments
    const int rA = tid >> 2;                 // A: row 0..63
    const int cA = (tid & 3) * 8;            // A: k-group of 8
    int rowA = m0 + rA; if (rowA > M - 1) rowA = M - 1;
    const float* aBase = &A[(size_t)rowA * K + cA];
    const int paA = kpos(cA);

    const int nB = tid & 63;                 // B: col 0..63
    const int kB = (tid >> 6) * 8;           // B: k-group of 8
    const float* wBase = &W[(size_t)kB * N + n0 + nB];
    const int pbB = kpos(kB);

    v8f acc0 = {0.f, 0.f, 0.f, 0.f, 0.f, 0.f, 0.f, 0.f};
    v8f acc1 = {0.f, 0.f, 0.f, 0.f, 0.f, 0.f, 0.f, 0.f};

    // prologue: tile 0 -> registers
    float4 ra0 = *(const float4*)aBase;
    float4 ra1 = *(const float4*)(aBase + 4);
    float rb[8];
#pragma unroll
    for (int j = 0; j < 8; ++j) rb[j] = wBase[(size_t)j * N];

    for (int k0 = 0, it = 0; k0 < K; k0 += 32, ++it) {
        const int cur = it & 1;
        // store previously loaded registers into LDS (fp32 -> bf16x2 packs, permuted)
        *(unsigned long long*)&As[cur][rA][paA]     = pack4bf(ra0.x, ra0.y, ra0.z, ra0.w);
        *(unsigned long long*)&As[cur][rA][paA + 4] = pack4bf(ra1.x, ra1.y, ra1.z, ra1.w);
        *(unsigned long long*)&Bs[cur][nB][pbB]     = pack4bf(rb[0], rb[1], rb[2], rb[3]);
        *(unsigned long long*)&Bs[cur][nB][pbB + 4] = pack4bf(rb[4], rb[5], rb[6], rb[7]);
        // issue global loads for the NEXT tile (latency hidden behind barrier+WMMA)
        if (k0 + 32 < K) {
            const float* ap = aBase + k0 + 32;
            ra0 = *(const float4*)ap;
            ra1 = *(const float4*)(ap + 4);
            const float* wp = wBase + (size_t)(k0 + 32) * N;
#pragma unroll
            for (int j = 0; j < 8; ++j) rb[j] = wp[(size_t)j * N];
            if (k0 + 64 < K) __builtin_prefetch(ap + 32, 0, 1);   // tile after next
        }
        __syncthreads();
        // contiguous 32B fragment loads (2x ds_load_b128 each)
        v16bf af  = *(const v16bf*)&As[cur][wm * 16 + l16][half * 16];
        v16bf bv0 = *(const v16bf*)&Bs[cur][wn * 32 + l16][half * 16];
        v16bf bv1 = *(const v16bf*)&Bs[cur][wn * 32 + 16 + l16][half * 16];
        acc0 = __builtin_amdgcn_wmma_f32_16x16x32_bf16(
            false, af, false, bv0, (short)0, acc0, false, false);
        acc1 = __builtin_amdgcn_wmma_f32_16x16x32_bf16(
            false, af, false, bv1, (short)0, acc1, false, false);
    }

    const int nA = n0 + wn * 32 + l16;
    const int nBcol = nA + 16;
    const float bva = bias ? bias[nA] : 0.0f;
    const float bvb = bias ? bias[nBcol] : 0.0f;
#pragma unroll
    for (int r = 0; r < 8; ++r) {
        int m = m0 + wm * 16 + half * 8 + r;   // C/D layout: VGPR r -> M=r / M=8+r
        if (m >= M) continue;
        float v0 = acc0[r] + bva;
        float v1 = acc1[r] + bvb;
        size_t off0 = (size_t)m * N + nA;
        size_t off1 = (size_t)m * N + nBcol;
        if (EPI == 0) {
            C[off0] = v0; C[off1] = v1;
        } else if (EPI == 1) {
            C[off0] = 0.5f * v0 * (1.0f + erff(v0 * 0.70710678118f));
            C[off1] = 0.5f * v1 * (1.0f + erff(v1 * 0.70710678118f));
        } else if (EPI == 2) {
            bool hit = (top1[m] == expert);
            C[off0] = hit ? v0 : 0.0f;
            C[off1] = hit ? v1 : 0.0f;
        } else {
            if (top1[m] == expert) { C[off0] += v0; C[off1] += v1; }
        }
    }
}

// --- per-(b,t) 8x8 head-mixing attention -----------------------------------
__global__ void head_attn(const float* __restrict__ Q, const float* __restrict__ K,
                          const float* __restrict__ V, float* __restrict__ O) {
    int gid = blockIdx.x * 256 + threadIdx.x;
    int n = gid >> 3, h = gid & 7;
    const float* q = Q + (size_t)n * 512 + h * 64;
    float sc[8];
#pragma unroll
    for (int g = 0; g < 8; ++g) {
        const float* kp = K + (size_t)n * 512 + g * 64;
        float s = 0.f;
        for (int d = 0; d < 64; ++d) s += q[d] * kp[d];
        sc[g] = s * 0.125f;                      // 64^-0.5
    }
    float mx = sc[0];
#pragma unroll
    for (int g = 1; g < 8; ++g) mx = fmaxf(mx, sc[g]);
    float sum = 0.f;
#pragma unroll
    for (int g = 0; g < 8; ++g) { sc[g] = expf(sc[g] - mx); sum += sc[g]; }
    float inv = 1.0f / sum;
#pragma unroll
    for (int g = 0; g < 8; ++g) sc[g] *= inv;
    float* o = O + (size_t)n * 512 + h * 64;
    for (int d = 0; d < 64; ++d) {
        float a = 0.f;
#pragma unroll
        for (int g = 0; g < 8; ++g) a += sc[g] * V[(size_t)n * 512 + g * 64 + d];
        o[d] = a;
    }
}

// --- LayerNorm over rows: out = LN(in1 (+ in2)) * g + b --------------------
__global__ void ln_rows(const float* __restrict__ in1, const float* __restrict__ in2,
                        const float* __restrict__ g, const float* __restrict__ b,
                        float* __restrict__ out, int Wd) {
    __shared__ float xbuf[576];
    __shared__ float red[256];
    __shared__ float stat[2];
    int row = blockIdx.x, tid = threadIdx.x;
    size_t base = (size_t)row * Wd;
    float s = 0.f;
    for (int i = tid; i < Wd; i += 256) {
        float v = in1[base + i] + (in2 ? in2[base + i] : 0.f);
        xbuf[i] = v; s += v;
    }
    red[tid] = s; __syncthreads();
    for (int off = 128; off > 0; off >>= 1) { if (tid < off) red[tid] += red[tid + off]; __syncthreads(); }
    if (tid == 0) stat[0] = red[0] / (float)Wd;
    __syncthreads();
    float mean = stat[0], vs = 0.f;
    for (int i = tid; i < Wd; i += 256) { float d = xbuf[i] - mean; vs += d * d; }
    red[tid] = vs; __syncthreads();
    for (int off = 128; off > 0; off >>= 1) { if (tid < off) red[tid] += red[tid + off]; __syncthreads(); }
    if (tid == 0) stat[1] = rsqrtf(red[0] / (float)Wd + 1e-5f);
    __syncthreads();
    float rstd = stat[1];
    for (int i = tid; i < Wd; i += 256)
        out[base + i] = (xbuf[i] - mean) * rstd * g[i] + b[i];
}

// --- MoE gate: softmax over 4 logits + top-1 (wave32 shuffle reduce) -------
__global__ void gate_kernel(const float* __restrict__ X, const float* __restrict__ gw,
                            float* __restrict__ gates4, int* __restrict__ top1) {
    int tok  = blockIdx.x * 8 + (threadIdx.x >> 5);
    int lane = threadIdx.x & 31;
    const float* x = X + (size_t)tok * 512;
    float a0 = 0.f, a1 = 0.f, a2 = 0.f, a3 = 0.f;
    for (int d = lane; d < 512; d += 32) {
        float xv = x[d];
        a0 += xv * gw[d * 4 + 0]; a1 += xv * gw[d * 4 + 1];
        a2 += xv * gw[d * 4 + 2]; a3 += xv * gw[d * 4 + 3];
    }
#pragma unroll
    for (int off = 16; off > 0; off >>= 1) {
        a0 += __shfl_xor(a0, off); a1 += __shfl_xor(a1, off);
        a2 += __shfl_xor(a2, off); a3 += __shfl_xor(a3, off);
    }
    if (lane == 0) {
        float mx = fmaxf(fmaxf(a0, a1), fmaxf(a2, a3));
        float e0 = expf(a0 - mx), e1 = expf(a1 - mx), e2 = expf(a2 - mx), e3 = expf(a3 - mx);
        float inv = 1.0f / (e0 + e1 + e2 + e3);
        gates4[tok * 4 + 0] = e0 * inv; gates4[tok * 4 + 1] = e1 * inv;
        gates4[tok * 4 + 2] = e2 * inv; gates4[tok * 4 + 3] = e3 * inv;
        int am = 0; float best = a0;                // first-max like jnp.argmax
        if (a1 > best) { best = a1; am = 1; }
        if (a2 > best) { best = a2; am = 2; }
        if (a3 > best) { best = a3; am = 3; }
        top1[tok] = am;
    }
}

// --- deterministic aux-loss reduction --------------------------------------
__global__ void aux_kernel(const float* __restrict__ gates4, const int* __restrict__ top1,
                           float* __restrict__ outAux) {
    __shared__ float red[256];
    __shared__ float tot[8];
    int tid = threadIdx.x;
    float imp[4] = {0, 0, 0, 0}, cnt[4] = {0, 0, 0, 0};
    for (int tok = tid; tok < 16384; tok += 256) {
        imp[0] += gates4[tok * 4 + 0]; imp[1] += gates4[tok * 4 + 1];
        imp[2] += gates4[tok * 4 + 2]; imp[3] += gates4[tok * 4 + 3];
        cnt[top1[tok]] += 1.0f;
    }
    for (int e = 0; e < 8; ++e) {
        red[tid] = (e < 4) ? imp[e] : cnt[e - 4];
        __syncthreads();
        for (int off = 128; off > 0; off >>= 1) { if (tid < off) red[tid] += red[tid + off]; __syncthreads(); }
        if (tid == 0) tot[e] = red[0];
        __syncthreads();
    }
    if (tid == 0) {
        float Nf = 16384.0f + 1e-9f, aux = 0.f;
        for (int e = 0; e < 4; ++e) aux += (tot[e] / Nf) * (tot[e + 4] / Nf);
        outAux[0] = 4.0f * aux;
    }
}

// --- attention pooling: s = tanh(x@W1+b1)@W2+b2 ----------------------------
__global__ void pool_score(const float* __restrict__ X2, const float* __restrict__ W1,
                           const float* __restrict__ b1, const float* __restrict__ W2,
                           const float* __restrict__ b2, float* __restrict__ S) {
    int row = blockIdx.x, j = threadIdx.x;     // 64 threads
    const float* x = X2 + (size_t)row * 512;
    float acc = b1[j];
    for (int d = 0; d < 512; ++d) acc += x[d] * W1[d * 64 + j];
    __shared__ float red[64];
    red[j] = tanhf(acc) * W2[j];
    __syncthreads();
    for (int off = 32; off > 0; off >>= 1) { if (j < off) red[j] += red[j + off]; __syncthreads(); }
    if (j == 0) S[row] = red[0] + b2[0];
}

// --- per-batch softmax over T and weighted pool ----------------------------
__global__ void pool_reduce(const float* __restrict__ X2, const float* __restrict__ S,
                            float* __restrict__ Mout) {
    int b = blockIdx.x, tid = threadIdx.x;
    __shared__ float w[1024];
    __shared__ float red[256];
    __shared__ float stat[2];
    float mx = -3.4e38f;
    for (int t = tid; t < 1024; t += 256) { float v = S[b * 1024 + t]; w[t] = v; mx = fmaxf(mx, v); }
    red[tid] = mx; __syncthreads();
    for (int off = 128; off > 0; off >>= 1) { if (tid < off) red[tid] = fmaxf(red[tid], red[tid + off]); __syncthreads(); }
    if (tid == 0) stat[0] = red[0];
    __syncthreads();
    mx = stat[0];
    float se = 0.f;
    for (int t = tid; t < 1024; t += 256) { float e = expf(w[t] - mx); w[t] = e; se += e; }
    red[tid] = se; __syncthreads();
    for (int off = 128; off > 0; off >>= 1) { if (tid < off) red[tid] += red[tid + off]; __syncthreads(); }
    if (tid == 0) stat[1] = 1.0f / red[0];
    __syncthreads();
    float inv = stat[1];
    for (int d = tid; d < 512; d += 256) {
        float acc = 0.f;
        for (int t = 0; t < 1024; ++t)
            acc += w[t] * X2[(size_t)(b * 1024 + t) * 512 + d];
        Mout[b * 512 + d] = acc * inv;
    }
}

// --- diffusion prep: cosine alpha-bar, m_t, timestep embedding, eps copy ---
__global__ void diff_prep(const float* __restrict__ Mv, const float* __restrict__ eps,
                          const int* __restrict__ tarr, float* __restrict__ Xin,
                          float* __restrict__ outEps) {
    int b = blockIdx.x, d = threadIdx.x;       // 576 threads
    int tv = tarr[b];
    const float PIH = 1.57079632679f;
    float c0 = cosf((0.008f / 1.008f) * PIH);
    float ang = (((float)tv / 1000.0f + 0.008f) / 1.008f) * PIH;
    float c = cosf(ang);
    float ab = (c * c) / (c0 * c0);
    ab = fminf(fmaxf(ab, 1e-5f), 0.99999f);
    float sa = sqrtf(ab), sb = sqrtf(1.0f - ab);
    if (d < 512) {
        float e = eps[b * 512 + d];
        Xin[b * 576 + d] = sa * Mv[b * 512 + d] + sb * e;
        outEps[b * 512 + d] = e;
    } else {
        int j = d - 512, jj = (j < 32) ? j : j - 32;
        float fr = expf(-logf(10000.0f) * (float)jj / 31.0f);
        float a = (float)tv * fr;
        Xin[b * 576 + d] = (j < 32) ? sinf(a) : cosf(a);
    }
}

// ---------------------------------------------------------------------------
extern "C" void kernel_launch(void* const* d_in, const int* in_sizes, int n_in,
                              void* d_out, int out_size, void* d_ws, size_t ws_size,
                              hipStream_t stream) {
    const float* a      = (const float*)d_in[0];
    const float* bb     = (const float*)d_in[1];
    const float* Wq     = (const float*)d_in[2];
    const float* Wk     = (const float*)d_in[3];
    const float* Wv     = (const float*)d_in[4];
    const float* Wp     = (const float*)d_in[5];
    const float* ca_g   = (const float*)d_in[6];
    const float* ca_b   = (const float*)d_in[7];
    const float* gateW  = (const float*)d_in[8];
    const float* moe_lng= (const float*)d_in[9];
    const float* moe_lnb= (const float*)d_in[10];
    const float* W1     = (const float*)d_in[11];
    const float* b1     = (const float*)d_in[12];
    const float* W2     = (const float*)d_in[13];
    const float* b2     = (const float*)d_in[14];
    const float* fuse_g = (const float*)d_in[15];
    const float* fuse_b = (const float*)d_in[16];
    const float* att_W1 = (const float*)d_in[17];
    const float* att_b1 = (const float*)d_in[18];
    const float* att_W2 = (const float*)d_in[19];
    const float* att_b2 = (const float*)d_in[20];
    const float* dh_g   = (const float*)d_in[21];
    const float* dh_b   = (const float*)d_in[22];
    const float* dh_W1  = (const float*)d_in[23];
    const float* dh_b1  = (const float*)d_in[24];
    const float* dh_W2  = (const float*)d_in[25];
    const float* dh_b2  = (const float*)d_in[26];
    const float* eps    = (const float*)d_in[27];
    const int*   tarr   = (const int*)d_in[28];
    float* out = (float*)d_out;

    const int Nt = 16384, D = 512, HID = 1024;
    float* ws = (float*)d_ws;
    const size_t BUF = (size_t)Nt * D;            // 8,388,608 floats
    float* Qb = ws;                               // Q, then proj, then y
    float* Kb = ws + BUF;                         // K, then x (post-CA LN)
    float* Vb = ws + 2 * BUF;                     // V, then per-expert LN, then x2
    float* Ab = ws + 3 * BUF;                     // attention output
    float* H1 = ws + 4 * BUF;                     // MoE hidden [Nt,1024]
    float* Sc = ws + 4 * BUF + (size_t)Nt * HID;  // pool scores [Nt]
    float* Mv = Sc + Nt;                          // pooled [16,512]
    float* Xin = Mv + 8192;                       // [16,576]
    float* Hd  = Xin + 9216;                      // [16,576]
    float* G1  = Hd + 9216;                       // [16,1024]
    float* gates4 = G1 + 16384;                   // [Nt,4]
    int*   top1 = (int*)(gates4 + (size_t)Nt * 4);

    dim3 blk(256);

    // Cross-attention block
    gemm_wmma<0><<<dim3(D/64, Nt/64), blk, 0, stream>>>(a,  Wq, nullptr, Qb, Nt, D, D, nullptr, 0);
    gemm_wmma<0><<<dim3(D/64, Nt/64), blk, 0, stream>>>(bb, Wk, nullptr, Kb, Nt, D, D, nullptr, 0);
    gemm_wmma<0><<<dim3(D/64, Nt/64), blk, 0, stream>>>(bb, Wv, nullptr, Vb, Nt, D, D, nullptr, 0);
    head_attn<<<dim3(Nt * 8 / 256), blk, 0, stream>>>(Qb, Kb, Vb, Ab);
    gemm_wmma<0><<<dim3(D/64, Nt/64), blk, 0, stream>>>(Ab, Wp, nullptr, Qb, Nt, D, D, nullptr, 0);
    ln_rows<<<dim3(Nt), blk, 0, stream>>>(Qb, a, ca_g, ca_b, Kb, D);          // x -> Kb

    // Switch-MoE
    gate_kernel<<<dim3(Nt / 8), blk, 0, stream>>>(Kb, gateW, gates4, top1);
    for (int e = 0; e < 4; ++e) {
        ln_rows<<<dim3(Nt), blk, 0, stream>>>(Kb, nullptr, moe_lng + e * D, moe_lnb + e * D, Vb, D);
        gemm_wmma<1><<<dim3(HID/64, Nt/64), blk, 0, stream>>>(
            Vb, W1 + (size_t)e * D * HID, b1 + e * HID, H1, Nt, HID, D, nullptr, 0);
        if (e == 0)
            gemm_wmma<2><<<dim3(D/64, Nt/64), blk, 0, stream>>>(
                H1, W2 + (size_t)e * HID * D, b2 + e * D, Qb, Nt, D, HID, top1, e);
        else
            gemm_wmma<3><<<dim3(D/64, Nt/64), blk, 0, stream>>>(
                H1, W2 + (size_t)e * HID * D, b2 + e * D, Qb, Nt, D, HID, top1, e);
    }
    ln_rows<<<dim3(Nt), blk, 0, stream>>>(Qb, Kb, fuse_g, fuse_b, Vb, D);     // x2 -> Vb

    // Attention pooling
    pool_score<<<dim3(Nt), dim3(64), 0, stream>>>(Vb, att_W1, att_b1, att_W2, att_b2, Sc);
    pool_reduce<<<dim3(16), blk, 0, stream>>>(Vb, Sc, Mv);

    // Diffusion head
    diff_prep<<<dim3(16), dim3(576), 0, stream>>>(Mv, eps, tarr, Xin, out + 8192);
    ln_rows<<<dim3(16), blk, 0, stream>>>(Xin, nullptr, dh_g, dh_b, Hd, 576);
    gemm_wmma<1><<<dim3(HID/64, 1), blk, 0, stream>>>(Hd, dh_W1, dh_b1, G1, 16, HID, 576, nullptr, 0);
    gemm_wmma<0><<<dim3(D/64, 1), blk, 0, stream>>>(G1, dh_W2, dh_b2, out, 16, D, HID, nullptr, 0);

    // aux scalar
    aux_kernel<<<dim3(1), blk, 0, stream>>>(gates4, top1, out + 16384);
}
